// AttentiveFP_mol_20469814133050
// MI455X (gfx1250) — compile-verified, matrix operand order checked by reading, not run on hardware
//
#include <hip/hip_runtime.h>
#include <cstdint>

#define Bsz   1024
#define Lsz   128
#define FPsz  256
#define NROWS (Bsz * Lsz)          /* 131072 */
#define NEGV  (-900000000.0f)
#define BNEPS 1e-6f

typedef __attribute__((ext_vector_type(16))) __bf16 v16bf;
typedef __attribute__((ext_vector_type(8)))  __bf16 bf16x8;
typedef __attribute__((ext_vector_type(8)))  float  v8f;

// TDM descriptor vector types (6-arg clang-23 builtin form)
typedef __attribute__((ext_vector_type(4))) unsigned int u32x4;
typedef __attribute__((ext_vector_type(8))) int          i32x8;
typedef __attribute__((ext_vector_type(4))) int          i32x4;

#if __has_builtin(__builtin_amdgcn_tensor_load_to_lds) && \
    __has_builtin(__builtin_amdgcn_s_wait_tensorcnt)
#define TDM_OK 1
#else
#define TDM_OK 0
#endif

// ---------------------------------------------------------------------------
// WMMA helpers (CDNA5 wave32, V_WMMA_F32_16X16X32_BF16)
// 16-bit A 16x32 layout: lane m (0-15) holds K = 0..7 and 16..23,
// lane m+16 holds K = 8..15 and 24..31 for row m.  B mirrors with lane = N,
// which for X @ W^T means each lane reads a contiguous row of W.
// ---------------------------------------------------------------------------
__device__ __forceinline__ v16bf load_frag(const __bf16* __restrict__ rowPtr, int k0) {
  bf16x8 lo = *(const bf16x8*)(rowPtr + k0);
  bf16x8 hi = *(const bf16x8*)(rowPtr + k0 + 16);
  v16bf r;
#pragma unroll
  for (int i = 0; i < 8; ++i) { r[i] = lo[i]; r[i + 8] = hi[i]; }
  return r;
}

__device__ __forceinline__ v8f wmma_bf16(v16bf a, v16bf b, v8f c) {
  return __builtin_amdgcn_wmma_f32_16x16x32_bf16(false, a, false, b, (short)0, c,
                                                 false, false);
}

__device__ __forceinline__ float wave_sum(float v) {
#pragma unroll
  for (int off = 16; off > 0; off >>= 1) v += __shfl_xor(v, off, 32);
  return v;
}
__device__ __forceinline__ float wave_max(float v) {
#pragma unroll
  for (int off = 16; off > 0; off >>= 1) v = fmaxf(v, __shfl_xor(v, off, 32));
  return v;
}

#if TDM_OK
// Issue a flat 1-D TDM copy of nelem bf16 elements global -> LDS.
// D# group layout per CDNA5 ISA ch.8 (count=1, type=2 "image", data_size=2B).
__device__ __forceinline__ void tdm_load_bf16(unsigned lds_off,
                                              const void* gsrc,
                                              unsigned nelem) {
  unsigned long long ga = (unsigned long long)(uintptr_t)gsrc;
  u32x4 g0;
  g0[0] = 1u;                                            // count=1, user mode
  g0[1] = lds_off;                                       // lds_addr [63:32]
  g0[2] = (unsigned)(ga & 0xFFFFFFFFu);                  // global_addr lo
  g0[3] = (unsigned)((ga >> 32) & 0x01FFFFFFu)           // global_addr[56:32]
          | (2u << 30);                                  // type = 2
  i32x8 g1;
  g1[0] = 0x00010000;                                    // data_size = 2 bytes
  g1[1] = (int)((nelem & 0xFFFFu) << 16);                // tensor_dim0 lo16
  g1[2] = (int)(((nelem >> 16) & 0xFFFFu)                // tensor_dim0 hi16
          | (1u << 16));                                 // tensor_dim1 = 1
  g1[3] = (int)((nelem & 0xFFFFu) << 16);                // tile_dim0 = nelem
  g1[4] = 0;                                             // tile_dim1/2 unused
  g1[5] = (int)nelem;                                    // dim0 stride lo32
  g1[6] = 0;                                             // stride hi / dim1 stride
  g1[7] = 0;
  i32x4 g2 = {0, 0, 0, 0};
  i32x4 g3 = {0, 0, 0, 0};
  i32x8 g4 = {0, 0, 0, 0, 0, 0, 0, 0};
  __builtin_amdgcn_tensor_load_to_lds(g0, g1, g2, g3, g4, 0);
}
#endif

// ---------------------------------------------------------------------------
// Init helpers
// ---------------------------------------------------------------------------
__global__ void zero2_kernel(float* a, float* b, int n) {
  int i = blockIdx.x * blockDim.x + threadIdx.x;
  if (i < n) { a[i] = 0.0f; b[i] = 0.0f; }
}

__global__ void f32_to_bf16_kernel(const float* __restrict__ src,
                                   __bf16* __restrict__ dst, int n) {
  int i = blockIdx.x * blockDim.x + threadIdx.x;
  if (i < n) dst[i] = (__bf16)src[i];
}

// ---------------------------------------------------------------------------
// Pass 1: per molecule b — masked sum -> super_FP / act, fp32 -> bf16 convert.
// ---------------------------------------------------------------------------
__global__ void prep_kernel(const float* __restrict__ atom,
                            const float* __restrict__ mask,
                            __bf16* __restrict__ X16,
                            float* __restrict__ superF,
                            float* __restrict__ act,
                            __bf16* __restrict__ h16) {
  int b = blockIdx.x;
  int k = threadIdx.x;
  __shared__ float mc[Lsz];
  if (k < Lsz) mc[k] = mask[b * Lsz + k];
  __syncthreads();
  const float* row = atom + (size_t)b * Lsz * FPsz;
  __bf16* xrow = X16 + (size_t)b * Lsz * FPsz;
  float acc = 0.0f;
#pragma unroll 4
  for (int l = 0; l < Lsz; ++l) {
    float x = row[l * FPsz + k];
    acc += x * mc[l];
    xrow[l * FPsz + k] = (__bf16)x;
  }
  superF[b * FPsz + k] = acc;
  act[b * FPsz + k] = acc > 0.0f ? acc : 0.0f;
  h16[b * FPsz + k] = (__bf16)acc;
}

// ---------------------------------------------------------------------------
// Pass 2: s2[b,l] = atom_FP[b,l,:] . Wf  plus per-b moments C1, C2.
// ---------------------------------------------------------------------------
__global__ void s2_kernel(const __bf16* __restrict__ X16,
                          const float* __restrict__ alignW,   // [512], Wf = +256
                          const float* __restrict__ alignb,
                          float* __restrict__ s2g,
                          float* __restrict__ C1, float* __restrict__ C2) {
  int b = blockIdx.x;
  int wave = threadIdx.x >> 5, lane = threadIdx.x & 31;
  __shared__ float s2s[Lsz];
  float wf[8];
#pragma unroll
  for (int i = 0; i < 8; ++i) wf[i] = alignW[FPsz + lane * 8 + i];
  for (int t = 0; t < 16; ++t) {
    int l = wave * 16 + t;
    const __bf16* xr = X16 + (size_t)(b * Lsz + l) * FPsz + lane * 8;
    float d = 0.0f;
#pragma unroll
    for (int i = 0; i < 8; ++i) d += (float)xr[i] * wf[i];
    d = wave_sum(d);
    if (lane == 0) s2s[l] = d;
  }
  __syncthreads();
  if (threadIdx.x < Lsz) s2g[b * Lsz + threadIdx.x] = s2s[threadIdx.x];
  if (threadIdx.x == 0) {
    float ab = alignb[0];
    float c1 = 0.0f, c2 = 0.0f;
    for (int l = 0; l < Lsz; ++l) {
      float c = s2s[l] + ab;
      c1 += c; c2 += c * c;
    }
    C1[b] = c1; C2[b] = c2;
  }
}

// ---------------------------------------------------------------------------
// Pass 3: streaming WMMA GEMM raw_xt = X16 @ Wt^T + bt, reduced on the fly to
// per-channel sum / sumsq (xt itself is never materialized).
// ---------------------------------------------------------------------------
__global__ void xt_stats_gemm(const __bf16* __restrict__ X16,
                              const __bf16* __restrict__ Wt16,
                              const float* __restrict__ attb,
                              float* __restrict__ colsum,
                              float* __restrict__ colsumsq) {
  int coltile = blockIdx.x & 15;
  int chunk   = blockIdx.x >> 4;
  int wave = threadIdx.x >> 5, lane = threadIdx.x & 31;
  int hf = lane >> 4, lr = lane & 15;

  const __bf16* wrow = Wt16 + (size_t)(coltile * 16 + lr) * FPsz;
  v16bf bfrag[8];
#pragma unroll
  for (int kb = 0; kb < 8; ++kb) bfrag[kb] = load_frag(wrow, kb * 32 + hf * 8);

  float btv = attb[coltile * 16 + lr];
  float ss = 0.0f, sq = 0.0f;
  int rt0 = chunk * 256 + wave * 64;
  for (int t = 0; t < 64; ++t) {
    const __bf16* arow = X16 + (size_t)((rt0 + t) * 16 + lr) * FPsz;
    v8f c = {};
#pragma unroll
    for (int kb = 0; kb < 8; ++kb) {
      v16bf a = load_frag(arow, kb * 32 + hf * 8);
      c = wmma_bf16(a, bfrag[kb], c);
    }
#pragma unroll
    for (int i = 0; i < 8; ++i) {
      float e = c[i] + btv;
      ss += e; sq += e * e;
    }
  }
  __shared__ float sS[16], sQ[16];
  if (threadIdx.x < 16) { sS[threadIdx.x] = 0.0f; sQ[threadIdx.x] = 0.0f; }
  __syncthreads();
  atomicAdd(&sS[lr], ss);
  atomicAdd(&sQ[lr], sq);
  __syncthreads();
  if (threadIdx.x < 16) {
    atomicAdd(&colsum[coltile * 16 + threadIdx.x], sS[threadIdx.x]);
    atomicAdd(&colsumsq[coltile * 16 + threadIdx.x], sQ[threadIdx.x]);
  }
}

// Fold BN(attend) into per-channel scale rg[f] and A-coefficient T[f].
__global__ void xt_fold_kernel(const float* __restrict__ colsum,
                               const float* __restrict__ colsumsq,
                               const float* __restrict__ attb,
                               const float* __restrict__ attG,
                               const float* __restrict__ attB,
                               float* __restrict__ rg, float* __restrict__ Tf) {
  int f = threadIdx.x;
  float m = colsum[f] * (1.0f / (float)NROWS);
  float var = colsumsq[f] * (1.0f / (float)NROWS) - m * m;
  float rstd = rsqrtf(var + BNEPS);
  float r = rstd * attG[f];
  rg[f] = r;
  Tf[f] = (attb[f] - m) * r + attB[f];
}

// ---------------------------------------------------------------------------
// Per layer: s1[b] = act[b,:] . Wa
// ---------------------------------------------------------------------------
__global__ void s1_kernel(const float* __restrict__ act,
                          const float* __restrict__ alignW,
                          float* __restrict__ s1) {
  int wave = threadIdx.x >> 5, lane = threadIdx.x & 31;
  int b = blockIdx.x * 8 + wave;
  const float* ar = act + (size_t)b * FPsz + lane * 8;
  float d = 0.0f;
#pragma unroll
  for (int i = 0; i < 8; ++i) d += ar[i] * alignW[lane * 8 + i];
  d = wave_sum(d);
  if (lane == 0) s1[b] = d;
}

// Global score-BN moments from per-b moments: single block of 1024.
__global__ void score_stats_kernel(const float* __restrict__ s1,
                                   const float* __restrict__ C1,
                                   const float* __restrict__ C2,
                                   float* __restrict__ stats) {
  int t = threadIdx.x;
  float s = s1[t];
  float c1 = C1[t], c2 = C2[t];
  float sumT = (float)Lsz * s + c1;
  float sqT  = (float)Lsz * s * s + 2.0f * s * c1 + c2;
  __shared__ float rA[1024], rB[1024];
  rA[t] = sumT; rB[t] = sqT;
  __syncthreads();
  for (int st = 512; st > 0; st >>= 1) {
    if (t < st) { rA[t] += rA[t + st]; rB[t] += rB[t + st]; }
    __syncthreads();
  }
  if (t == 0) {
    float mean = rA[0] * (1.0f / (float)NROWS);
    float var  = rB[0] * (1.0f / (float)NROWS) - mean * mean;
    stats[0] = mean;
    stats[1] = rsqrtf(var + BNEPS);
  }
}

// ---------------------------------------------------------------------------
// Attention softmax + weighted context wctx[b,k] = sum_l attn[b,l]*atom[b,l,k]
// grid = B, block = 256 (8 waves).  The molecule's [128,256] bf16 tile is
// staged into LDS with TENSOR_LOAD_TO_LDS (double-buffered, 4 stages of 32
// rows), overlapping the TDM DMA with the softmax math; the reduction then
// runs out of LDS instead of issuing 128 strided global loads per thread.
// ---------------------------------------------------------------------------
__global__ void attn_kernel(const __bf16* __restrict__ X16,
                            const float* __restrict__ mask,
                            const float* __restrict__ s2g,
                            const float* __restrict__ s1,
                            const float* __restrict__ stats,
                            const float* __restrict__ alignb,
                            const float* __restrict__ alignG,
                            const float* __restrict__ alignB,
                            __bf16* __restrict__ wctx16,
                            float* __restrict__ Avec) {
  int b = blockIdx.x;
  int tid = threadIdx.x;
  int wave = tid >> 5, lane = tid & 31;
  __shared__ float sc_s[Lsz], e_s[Lsz], a_s[Lsz], wred[8];
  __shared__ float gmax_s, gsum_s, asum_s;
#if TDM_OK
  __shared__ __bf16 Xs[2][32 * FPsz];          // 2 x 16 KB double buffer
  const __bf16* gsrc = X16 + (size_t)b * Lsz * FPsz;
  unsigned lds_base = (unsigned)(uintptr_t)(void*)&Xs[0][0];
  if (tid < 32) {                              // wave 0 drives the TDM
    tdm_load_bf16(lds_base, gsrc, 32 * FPsz);  // stage 0 in flight
  }
#endif

  float mean = stats[0], rstd = stats[1];
  float s1v = s1[b], ab = alignb[0], gam = alignG[0], bet = alignB[0];

  if (tid < Lsz) {
    float pre = s1v + s2g[b * Lsz + tid] + ab;
    float bn = (pre - mean) * rstd * gam + bet;
    float ly = bn > 0.0f ? bn : 0.01f * bn;
    float mk = mask[b * Lsz + tid];
    sc_s[tid] = ly + (mk == 0.0f ? NEGV : 0.0f);
    a_s[tid] = mk;                         // stash mask
  }
  __syncthreads();
  // max over 128 (duplicated reads are harmless for max)
  float v = sc_s[tid & (Lsz - 1)];
  v = wave_max(v);
  if (lane == 0) wred[wave] = v;
  __syncthreads();
  if (tid == 0) {
    float m = wred[0];
    for (int i = 1; i < 8; ++i) m = fmaxf(m, wred[i]);
    gmax_s = m;
  }
  __syncthreads();
  if (tid < Lsz) {
    float e = __expf(sc_s[tid] - gmax_s);
    e_s[tid] = e;
    a_s[tid] = e * a_s[tid];               // e * mask
  }
  __syncthreads();
  float ve = (tid < Lsz) ? e_s[tid] : 0.0f;
  float va = (tid < Lsz) ? a_s[tid] : 0.0f;
  ve = wave_sum(ve);
  va = wave_sum(va);
  if (lane == 0) wred[wave] = ve;
  __syncthreads();
  if (tid == 0) { float s = 0; for (int i = 0; i < 8; ++i) s += wred[i]; gsum_s = s; }
  __syncthreads();
  if (lane == 0) wred[wave] = va;
  __syncthreads();
  if (tid == 0) { float s = 0; for (int i = 0; i < 8; ++i) s += wred[i]; asum_s = s; }
  __syncthreads();
  float inv = 1.0f / gsum_s;
  if (tid < Lsz) sc_s[tid] = a_s[tid] * inv;   // attn[l]
  if (tid == 0) Avec[b] = asum_s * inv;        // sum_l attn
  __syncthreads();

  float acc = 0.0f;
#if TDM_OK
#pragma unroll
  for (int s = 0; s < 4; ++s) {
    if (tid < 32) {
      if (s < 3) {
        tdm_load_bf16(lds_base + (unsigned)(((s + 1) & 1) * (32 * FPsz * 2)),
                      gsrc + (size_t)(s + 1) * 32 * FPsz, 32 * FPsz);
        __builtin_amdgcn_s_wait_tensorcnt(1);   // stage s complete (in-order)
      } else {
        __builtin_amdgcn_s_wait_tensorcnt(0);
      }
    }
    __syncthreads();
    const __bf16* bp = &Xs[s & 1][0];
#pragma unroll 4
    for (int l = 0; l < 32; ++l)
      acc += sc_s[s * 32 + l] * (float)bp[l * FPsz + tid];
    __syncthreads();                            // buffer reuse fence
  }
#else
  const __bf16* xp = X16 + (size_t)b * Lsz * FPsz + tid;
#pragma unroll 4
  for (int l = 0; l < Lsz; ++l) acc += sc_s[l] * (float)xp[l * FPsz];
#endif
  wctx16[(size_t)b * FPsz + tid] = (__bf16)acc;
}

// ---------------------------------------------------------------------------
// ctx = elu( (wctx @ Wt^T) * rg + A * T )  via WMMA; output bf16 for GRU GEMM.
// ---------------------------------------------------------------------------
__global__ void ctx_gemm_kernel(const __bf16* __restrict__ wctx16,
                                const __bf16* __restrict__ Wt16,
                                const float* __restrict__ rg,
                                const float* __restrict__ Tf,
                                const float* __restrict__ Avec,
                                __bf16* __restrict__ ctx16) {
  int tile = blockIdx.x * 4 + (threadIdx.x >> 5);
  int rowtile = tile >> 4, coltile = tile & 15;
  int lane = threadIdx.x & 31, hf = lane >> 4, lr = lane & 15;
  const __bf16* wrow = Wt16 + (size_t)(coltile * 16 + lr) * FPsz;
  const __bf16* arow = wctx16 + (size_t)(rowtile * 16 + lr) * FPsz;
  v8f c = {};
#pragma unroll
  for (int kb = 0; kb < 8; ++kb) {
    v16bf a = load_frag(arow, kb * 32 + hf * 8);
    v16bf bm = load_frag(wrow, kb * 32 + hf * 8);
    c = wmma_bf16(a, bm, c);
  }
  int f = coltile * 16 + lr;
  float rgv = rg[f], Tv = Tf[f];
#pragma unroll
  for (int i = 0; i < 8; ++i) {
    int row = rowtile * 16 + hf * 8 + i;
    float vv = c[i] * rgv + Avec[row] * Tv;
    vv = vv > 0.0f ? vv : (__expf(vv) - 1.0f);   // ELU
    ctx16[(size_t)row * FPsz + f] = (__bf16)vv;
  }
}

// ---------------------------------------------------------------------------
// GRU gate GEMM: out[1024,768] = U16 @ W16^T + bias  (WMMA, 64x48 tile grid)
// ---------------------------------------------------------------------------
__global__ void gru_gemm_kernel(const __bf16* __restrict__ U16,
                                const __bf16* __restrict__ W16,
                                const float* __restrict__ bias,
                                float* __restrict__ out) {
  int tile = blockIdx.x * 4 + (threadIdx.x >> 5);
  int rowtile = tile / 48, coltile = tile % 48;
  int lane = threadIdx.x & 31, hf = lane >> 4, lr = lane & 15;
  const __bf16* wrow = W16 + (size_t)(coltile * 16 + lr) * FPsz;
  const __bf16* arow = U16 + (size_t)(rowtile * 16 + lr) * FPsz;
  v8f c = {};
#pragma unroll
  for (int kb = 0; kb < 8; ++kb) {
    v16bf a = load_frag(arow, kb * 32 + hf * 8);
    v16bf bm = load_frag(wrow, kb * 32 + hf * 8);
    c = wmma_bf16(a, bm, c);
  }
  int col = coltile * 16 + lr;
  float bv = bias[col];
#pragma unroll
  for (int i = 0; i < 8; ++i) {
    int row = rowtile * 16 + hf * 8 + i;
    out[(size_t)row * 768 + col] = c[i] + bv;
  }
}

// GRU elementwise update; writes super_FP / act into d_out, h16 for next GEMM.
__global__ void gru_update_kernel(const float* __restrict__ gi,
                                  const float* __restrict__ gh,
                                  float* __restrict__ superF,
                                  float* __restrict__ act,
                                  __bf16* __restrict__ h16) {
  int b = blockIdx.x, f = threadIdx.x;
  size_t o = (size_t)b * 768;
  float ir = gi[o + f], iz = gi[o + 256 + f], in_ = gi[o + 512 + f];
  float hr = gh[o + f], hz = gh[o + 256 + f], hn = gh[o + 512 + f];
  float h = superF[(size_t)b * FPsz + f];
  float r = 1.0f / (1.0f + __expf(-(ir + hr)));
  float z = 1.0f / (1.0f + __expf(-(iz + hz)));
  float n = tanhf(in_ + r * hn);
  float nh = (1.0f - z) * n + z * h;
  superF[(size_t)b * FPsz + f] = nh;
  act[(size_t)b * FPsz + f] = nh > 0.0f ? nh : 0.0f;
  h16[(size_t)b * FPsz + f] = (__bf16)nh;
}

// ---------------------------------------------------------------------------
extern "C" void kernel_launch(void* const* d_in, const int* in_sizes, int n_in,
                              void* d_out, int out_size, void* d_ws, size_t ws_size,
                              hipStream_t stream) {
  (void)in_sizes; (void)n_in; (void)out_size; (void)ws_size;
  const float* atom   = (const float*)d_in[0];
  const float* mask   = (const float*)d_in[1];
  const float* alignW = (const float*)d_in[2];
  const float* alignb = (const float*)d_in[3];
  const float* alignG = (const float*)d_in[4];
  const float* alignB = (const float*)d_in[5];
  const float* attW   = (const float*)d_in[6];
  const float* attb   = (const float*)d_in[7];
  const float* attG   = (const float*)d_in[8];
  const float* attB   = (const float*)d_in[9];
  const float* Wih    = (const float*)d_in[10];
  const float* Whh    = (const float*)d_in[11];
  const float* bih    = (const float*)d_in[12];
  const float* bhh    = (const float*)d_in[13];

  float* out    = (float*)d_out;
  float* superF = out;
  float* act    = out + (size_t)Bsz * FPsz;

  char* ws = (char*)d_ws;
  size_t off = 0;
  auto alloc = [&](size_t bytes) -> void* {
    void* p = ws + off;
    off += (bytes + 255) & ~(size_t)255;
    return p;
  };
  __bf16* X16     = (__bf16*)alloc((size_t)NROWS * FPsz * 2);
  __bf16* Wt16    = (__bf16*)alloc((size_t)FPsz * FPsz * 2);
  __bf16* Wih16   = (__bf16*)alloc((size_t)768 * FPsz * 2);
  __bf16* Whh16   = (__bf16*)alloc((size_t)768 * FPsz * 2);
  __bf16* h16     = (__bf16*)alloc((size_t)Bsz * FPsz * 2);
  __bf16* wctx16  = (__bf16*)alloc((size_t)Bsz * FPsz * 2);
  __bf16* ctx16   = (__bf16*)alloc((size_t)Bsz * FPsz * 2);
  float* s2g      = (float*)alloc((size_t)NROWS * 4);
  float* C1       = (float*)alloc(Bsz * 4);
  float* C2       = (float*)alloc(Bsz * 4);
  float* colsum   = (float*)alloc(FPsz * 4);
  float* colsumsq = (float*)alloc(FPsz * 4);
  float* rg       = (float*)alloc(FPsz * 4);
  float* Tf       = (float*)alloc(FPsz * 4);
  float* s1       = (float*)alloc(Bsz * 4);
  float* stats    = (float*)alloc(2 * 4);
  float* Avec     = (float*)alloc(Bsz * 4);
  float* gi       = (float*)alloc((size_t)Bsz * 768 * 4);
  float* gh       = (float*)alloc((size_t)Bsz * 768 * 4);

  // init: zero stat accumulators, convert weights to bf16
  zero2_kernel<<<1, 256, 0, stream>>>(colsum, colsumsq, FPsz);
  f32_to_bf16_kernel<<<(FPsz * FPsz) / 256, 256, 0, stream>>>(attW, Wt16, FPsz * FPsz);
  f32_to_bf16_kernel<<<(768 * FPsz) / 256, 256, 0, stream>>>(Wih, Wih16, 768 * FPsz);
  f32_to_bf16_kernel<<<(768 * FPsz) / 256, 256, 0, stream>>>(Whh, Whh16, 768 * FPsz);

  // layer-invariant precompute
  prep_kernel<<<Bsz, 256, 0, stream>>>(atom, mask, X16, superF, act, h16);
  s2_kernel<<<Bsz, 256, 0, stream>>>(X16, alignW, alignb, s2g, C1, C2);
  xt_stats_gemm<<<512, 128, 0, stream>>>(X16, Wt16, attb, colsum, colsumsq);
  xt_fold_kernel<<<1, 256, 0, stream>>>(colsum, colsumsq, attb, attG, attB, rg, Tf);

  for (int layer = 0; layer < 3; ++layer) {
    s1_kernel<<<Bsz / 8, 256, 0, stream>>>(act, alignW, s1);
    score_stats_kernel<<<1, 1024, 0, stream>>>(s1, C1, C2, stats);
    attn_kernel<<<Bsz, 256, 0, stream>>>(X16, mask, s2g, s1, stats,
                                         alignb, alignG, alignB, wctx16, Avec);
    ctx_gemm_kernel<<<256, 128, 0, stream>>>(wctx16, Wt16, rg, Tf, Avec, ctx16);
    gru_gemm_kernel<<<768, 128, 0, stream>>>(ctx16, Wih16, bih, gi);
    gru_gemm_kernel<<<768, 128, 0, stream>>>(h16, Whh16, bhh, gh);
    gru_update_kernel<<<Bsz, 256, 0, stream>>>(gi, gh, superF, act, h16);
  }
}
